// ImageRNNSystem_37108517437491
// MI455X (gfx1250) — matrix operand order repaired and checked by reading, not compiled, not run on hardware
//
#include <hip/hip_runtime.h>

#define BATCH 4096
#define HID   256
#define NL    6
#define TT    49
#define MT    16              // batch rows per workgroup
#define HPAD  (HID + 8)       // LDS row padding (halves) to spread banks

typedef __bf16 bf16_t;
typedef bf16_t v16bf __attribute__((ext_vector_type(16)));
typedef float  v8f   __attribute__((ext_vector_type(8)));

union FragU { uint4 u[2]; v16bf v; };

__device__ __forceinline__ unsigned short f2bf(float f) {
    unsigned u = __float_as_uint(f);
    u += 0x7FFFu + ((u >> 16) & 1u);     // round-to-nearest-even
    return (unsigned short)(u >> 16);
}
__device__ __forceinline__ float bf2f(unsigned short h) {
    return __uint_as_float(((unsigned)h) << 16);
}

// Branchless tanh on native v_exp_f32 / v_rcp_f32:
//   tanh(x) = 1 - 2 / (exp2(2*log2(e)*x) + 1)
// Saturates to +/-1 for large |x| with no slow path or EXEC divergence.
__device__ __forceinline__ float fast_tanh(float x) {
    float e = __builtin_amdgcn_exp2f(x * 2.8853900817779268f);
    return 1.0f - 2.0f * __builtin_amdgcn_rcpf(e + 1.0f);
}

// ---------------------------------------------------------------------------
// Kernel 1: xf[b][t] = sum_j x[b][j] * W_p1[t][j]       (4096 x 49)
// ---------------------------------------------------------------------------
__global__ __launch_bounds__(256) void proj_kernel(
    const float* __restrict__ x,      // [B][196]
    const float* __restrict__ W_p1,   // [49][196]
    float* __restrict__ xf)           // [B][49]
{
    int idx = blockIdx.x * blockDim.x + threadIdx.x;
    if (idx >= BATCH * TT) return;
    int b = idx / TT, t = idx % TT;
    const float* xr = x + (size_t)b * 196;
    const float* wr = W_p1 + (size_t)t * 196;
    float s = 0.f;
    #pragma unroll 4
    for (int j = 0; j < 196; ++j) s = fmaf(xr[j], wr[j], s);
    xf[idx] = s;
}

// ---------------------------------------------------------------------------
// Kernel 2: pack W_in (5 mats) + W_hh (6 mats) f32 -> bf16 WMMA B-fragments.
// Fragment (kt, nt): 32 lanes x 16 contiguous halves per lane.
//   lane = (n&15) + ((kin>>3)&1)*16, element j = (kin&7) + (kin>=16 ? 8 : 0), K = kt*32+kin.
// ---------------------------------------------------------------------------
__global__ __launch_bounds__(256) void pack_kernel(
    const float* __restrict__ W_in,   // [5][256][256]
    const float* __restrict__ W_hh,   // [6][256][256]
    unsigned short* __restrict__ wpack)
{
    int id = blockIdx.x * blockDim.x + threadIdx.x;
    if (id >= 11 * HID * HID) return;
    int mat = id / (HID * HID);
    int rem = id % (HID * HID);
    int k = rem / HID;
    int n = rem % HID;
    const float* src = (mat < 5) ? (W_in + (size_t)mat * HID * HID)
                                 : (W_hh + (size_t)(mat - 5) * HID * HID);
    float v = src[(size_t)k * HID + n];
    int kt   = k >> 5;
    int kin  = k & 31;
    int j    = (kin & 7) + ((kin >= 16) ? 8 : 0);
    int lane = (n & 15) + (((kin >> 3) & 1) ? 16 : 0);
    int nt   = n >> 4;
    size_t dst = (size_t)mat * HID * HID +
                 ((((size_t)kt * 16 + nt) * 32 + lane) * 16 + j);
    wpack[dst] = f2bf(v);
}

// ---------------------------------------------------------------------------
// Kernel 3: persistent per-batch-tile RNN. 256 blocks x 256 threads (8 waves).
// Hidden state for all 6 layers lives in LDS (bf16) for the whole sequence.
// ---------------------------------------------------------------------------
__global__ __launch_bounds__(256) void rnn_kernel(
    const float* __restrict__ xf,              // [B][49]
    const unsigned short* __restrict__ wpack,  // packed bf16 weights (11 mats)
    const float* __restrict__ W_in0,           // [256]
    const float* __restrict__ b_rnn,           // [6][256]
    const float* __restrict__ W_fc1,           // [8][256]
    const float* __restrict__ W_fc2,           // [10][8]
    const float* __restrict__ b_fc2,           // [10]
    float* __restrict__ out)                   // [B][10]
{
    __shared__ unsigned short hs[NL][MT][HPAD];   // ~51 KB bf16 hidden state
    __shared__ float sBias[NL * HID];
    __shared__ float sW0[HID];
    __shared__ float sX[MT];
    __shared__ float sR1[MT][8];

    const int tid     = threadIdx.x;
    const int lane    = tid & 31;
    const int wave    = tid >> 5;
    const int rowBase = blockIdx.x * MT;

    const int nlo   = lane & 15;            // N within tile (C/D layout)
    const int mhalf = (lane >> 4) << 3;     // 0 | 8 : M-half (C/D) and K-group (A)

    for (int i = tid; i < NL * MT * HPAD; i += 256)
        ((unsigned short*)hs)[i] = 0;       // cells start at zero
    for (int i = tid; i < NL * HID; i += 256) sBias[i] = b_rnn[i];
    for (int i = tid; i < HID;      i += 256) sW0[i]   = W_in0[i];
    __syncthreads();

    v8f acc[2];

    for (int t = 0; t < TT; ++t) {
        if (tid < MT) sX[tid] = xf[(size_t)(rowBase + tid) * TT + t];
        __syncthreads();

        for (int layer = 0; layer < NL; ++layer) {
            // ---- compute phase: read-only on hs (uniform control flow) ----
            #pragma unroll
            for (int q = 0; q < 2; ++q) {
                const int nt = wave * 2 + q;
                const int n  = nt * 16 + nlo;
                const float bias = sBias[layer * HID + n];
                v8f a;
                #pragma unroll
                for (int r = 0; r < 8; ++r) a[r] = bias;

                if (layer == 0) {
                    // outer product: p[m] * W_in0[n]
                    const float w0n = sW0[n];
                    #pragma unroll
                    for (int r = 0; r < 8; ++r)
                        a[r] = fmaf(sX[mhalf + r], w0n, a[r]);
                } else {
                    // h_t^{layer-1} @ W_in[layer-1]
                    const unsigned short* wb =
                        wpack + (size_t)(layer - 1) * (HID * HID);
                    #pragma unroll
                    for (int kt = 0; kt < 8; ++kt) {
                        FragU fa, fb;
                        fa.u[0] = *(const uint4*)&hs[layer - 1][nlo][kt * 32 + mhalf];
                        fa.u[1] = *(const uint4*)&hs[layer - 1][nlo][kt * 32 + mhalf + 16];
                        const unsigned short* bp =
                            wb + (((size_t)kt * 16 + nt) * 32 + lane) * 16;
                        fb.u[0] = ((const uint4*)bp)[0];
                        fb.u[1] = ((const uint4*)bp)[1];
                        a = __builtin_amdgcn_wmma_f32_16x16x32_bf16(
                                false, fa.v, false, fb.v, (short)0, a, false, false);
                    }
                }
                // h_{t-1}^{layer} @ W_hh[layer]
                const unsigned short* wh =
                    wpack + (size_t)(5 + layer) * (HID * HID);
                #pragma unroll
                for (int kt = 0; kt < 8; ++kt) {
                    FragU fa, fb;
                    fa.u[0] = *(const uint4*)&hs[layer][nlo][kt * 32 + mhalf];
                    fa.u[1] = *(const uint4*)&hs[layer][nlo][kt * 32 + mhalf + 16];
                    const unsigned short* bp =
                        wh + (((size_t)kt * 16 + nt) * 32 + lane) * 16;
                    fb.u[0] = ((const uint4*)bp)[0];
                    fb.u[1] = ((const uint4*)bp)[1];
                    a = __builtin_amdgcn_wmma_f32_16x16x32_bf16(
                            false, fa.v, false, fb.v, (short)0, a, false, false);
                }
                acc[q] = a;
            }
            __syncthreads();   // all reads of hs[layer] done

            // ---- write phase: branchless tanh + bf16 store back to LDS ----
            #pragma unroll
            for (int q = 0; q < 2; ++q) {
                const int n = (wave * 2 + q) * 16 + nlo;
                #pragma unroll
                for (int r = 0; r < 8; ++r)
                    hs[layer][mhalf + r][n] = f2bf(fast_tanh(acc[q][r]));
            }
            __syncthreads();   // hs[layer] now holds h_t^{layer}
        }
    }

    // ---- FC head: relu(h @ W_fc1^T) @ W_fc2^T + b_fc2 (parallelized) ----
    if (tid < MT * 8) {                       // 128 threads: one (row, j) dot each
        const int row = tid >> 3;
        const int j   = tid & 7;
        const float* w1 = W_fc1 + (size_t)j * HID;
        float s = 0.f;
        for (int k = 0; k < HID; ++k)
            s = fmaf(bf2f(hs[NL - 1][row][k]), w1[k], s);
        sR1[row][j] = fmaxf(s, 0.f);
    }
    __syncthreads();
    if (tid < MT * 10) {                      // 160 threads: one (row, o) output each
        const int row = tid / 10;
        const int o   = tid % 10;
        float v = b_fc2[o];
        #pragma unroll
        for (int j = 0; j < 8; ++j)
            v = fmaf(sR1[row][j], W_fc2[o * 8 + j], v);
        out[(size_t)(rowBase + row) * 10 + o] = v;
    }
}

// ---------------------------------------------------------------------------
extern "C" void kernel_launch(void* const* d_in, const int* in_sizes, int n_in,
                              void* d_out, int out_size, void* d_ws, size_t ws_size,
                              hipStream_t stream) {
    const float* x      = (const float*)d_in[0];
    // d_in[1] = cell_out (zeros) -- not needed, hidden state zeroed in-kernel
    const float* W_p1   = (const float*)d_in[2];
    const float* W_in0  = (const float*)d_in[3];
    const float* W_in   = (const float*)d_in[4];
    const float* W_hh   = (const float*)d_in[5];
    const float* b_rnn  = (const float*)d_in[6];
    const float* W_fc1  = (const float*)d_in[7];
    const float* W_fc2  = (const float*)d_in[8];
    const float* b_fc2  = (const float*)d_in[9];
    float* out = (float*)d_out;

    float*          xf    = (float*)d_ws;                               // 4096*49*4 B
    unsigned short* wpack = (unsigned short*)((char*)d_ws +
                             (size_t)BATCH * TT * sizeof(float));        // 11*64K bf16

    proj_kernel<<<(BATCH * TT + 255) / 256, 256, 0, stream>>>(x, W_p1, xf);
    pack_kernel<<<(11 * HID * HID + 255) / 256, 256, 0, stream>>>(W_in, W_hh, wpack);
    rnn_kernel<<<BATCH / MT, 256, 0, stream>>>(xf, wpack, W_in0, b_rnn,
                                               W_fc1, W_fc2, b_fc2, out);
}
